// AdvancedGraphNeuralNetwork_2018634629266
// MI455X (gfx1250) — compile-verified, matrix operand order checked by reading, not run on hardware
//
#include <hip/hip_runtime.h>
#include <cmath>
#include <limits>
#include <cstdint>
#include <cstddef>

// ---------------------------------------------------------------------------
// AdvancedGraphNeuralNetwork for MI455X (gfx1250, wave32, WMMA).
// All dense GEMMs -> v_wmma_f32_16x16x32_bf16 (fp32 accum).
// Segment softmax / scatter via L2 float atomics (workload is L2-resident).
// EdgeConv MLP fused per 16-edge tile with LDS staging between the two GEMMs.
// ---------------------------------------------------------------------------

#define N_NODES 50000
#define N_EDGES 640000
#define D_IN    128
#define D_H     64
#define N_HEAD  4
#define HD      256              // N_HEAD * D_H
#define N_G     16
#define E_SL    (N_EDGES + N_NODES)   // edges + self loops (GAT)

typedef __bf16 v8bf  __attribute__((ext_vector_type(8)));
typedef __bf16 v16bf __attribute__((ext_vector_type(16)));
typedef float  v8f   __attribute__((ext_vector_type(8)));

// ---------------- atomics (agent scope, relaxed -> global_atomic_*) --------
__device__ __forceinline__ void atomAddF(float* p, float v) {
  __hip_atomic_fetch_add(p, v, __ATOMIC_RELAXED, __HIP_MEMORY_SCOPE_AGENT);
}
__device__ __forceinline__ void atomMaxF(float* p, float v) {
  if (v >= 0.f)
    __hip_atomic_fetch_max((int*)p, __float_as_int(v),
                           __ATOMIC_RELAXED, __HIP_MEMORY_SCOPE_AGENT);
  else
    __hip_atomic_fetch_min((unsigned int*)p, __float_as_uint(v),
                           __ATOMIC_RELAXED, __HIP_MEMORY_SCOPE_AGENT);
}

// ---------------- small utility kernels ------------------------------------
__global__ void k_fill(float* __restrict__ p, float v, size_t n) {
  size_t i = (size_t)blockIdx.x * blockDim.x + threadIdx.x;
  if (i < n) p[i] = v;
}

__global__ void k_f2bf(const float* __restrict__ in, __bf16* __restrict__ out, size_t n) {
  size_t i = (size_t)blockIdx.x * blockDim.x + threadIdx.x;
  if (i < n) out[i] = (__bf16)in[i];
}

// W [K][Nout] (row major, as in reference) -> Wt [Nout][K] bf16
__global__ void k_wT(const float* __restrict__ W, __bf16* __restrict__ Wt, int K, int Nout) {
  int i = blockIdx.x * blockDim.x + threadIdx.x;
  if (i >= K * Nout) return;
  int k = i / Nout, n = i % Nout;
  Wt[(size_t)n * K + k] = (__bf16)W[i];
}

// ---------------- WMMA GEMM: C[M,Nout] = A[M,K] @ Wt^T (+bias)(+Cadd)(relu) -
// A bf16 row-major, Wt bf16 [Nout][K]. M %16 == 0, K %32 == 0, Nout %64 == 0.
// Each wave: one 16x64 tile (4 col sub-tiles of 16).
__global__ void k_gemm(const __bf16* __restrict__ A, const __bf16* __restrict__ Wt,
                       const float* __restrict__ bias, const float* __restrict__ Cadd,
                       float* __restrict__ C, __bf16* __restrict__ Cbf,
                       int M, int K, int Nout, int relu) {
  int wave = (int)(((size_t)blockIdx.x * blockDim.x + threadIdx.x) >> 5);
  int lane = threadIdx.x & 31;
  int colTiles = Nout >> 6;
  int tiles = (M >> 4) * colTiles;
  if (wave >= tiles) return;
  int rowTile = wave / colTiles;
  int colGrp  = wave % colTiles;
  int ll  = lane & 15;   // A row / B col within tile
  int sel = lane >> 4;   // K-half select per ISA fragment layout
  int n0  = colGrp << 6;

  const __bf16* arow = A + (size_t)((rowTile << 4) + ll) * K;

  v8f acc[4] = {};
  for (int k0 = 0; k0 < K; k0 += 32) {
    v8bf alo = *(const v8bf*)(arow + k0 + sel * 8);
    v8bf ahi = *(const v8bf*)(arow + k0 + 16 + sel * 8);
    v16bf af = __builtin_shufflevector(alo, ahi, 0,1,2,3,4,5,6,7,8,9,10,11,12,13,14,15);
#pragma unroll
    for (int c = 0; c < 4; ++c) {
      const __bf16* brow = Wt + (size_t)(n0 + (c << 4) + ll) * K;
      v8bf blo = *(const v8bf*)(brow + k0 + sel * 8);
      v8bf bhi = *(const v8bf*)(brow + k0 + 16 + sel * 8);
      v16bf bfr = __builtin_shufflevector(blo, bhi, 0,1,2,3,4,5,6,7,8,9,10,11,12,13,14,15);
      acc[c] = __builtin_amdgcn_wmma_f32_16x16x32_bf16(false, af, false, bfr,
                                                       (short)0, acc[c], false, false);
    }
  }
#pragma unroll
  for (int c = 0; c < 4; ++c) {
    int n = n0 + (c << 4) + ll;
    float bb = bias ? bias[n] : 0.f;
#pragma unroll
    for (int r = 0; r < 8; ++r) {
      int row = (rowTile << 4) + r + sel * 8;
      size_t idx = (size_t)row * Nout + n;
      float v = acc[c][r] + bb;
      if (Cadd) v += Cadd[idx];
      if (relu) v = fmaxf(v, 0.f);
      C[idx] = v;
      if (Cbf) Cbf[idx] = (__bf16)v;
    }
  }
}

// ---------------- GAT attention helpers ------------------------------------
// ls[n,h] = <h[n,h,:], asrc[h,:]>,  ld likewise.
__global__ void k_attn_logits(const float* __restrict__ h,
                              const float* __restrict__ asrc, const float* __restrict__ adst,
                              float* __restrict__ ls, float* __restrict__ ld) {
  int i = blockIdx.x * blockDim.x + threadIdx.x;
  if (i >= N_NODES * N_HEAD) return;
  int nd = i >> 2, hh = i & 3;
  const float* hp = h + (size_t)nd * HD + hh * D_H;
  float s = 0.f, d = 0.f;
#pragma unroll 4
  for (int j = 0; j < D_H; ++j) { float v = hp[j]; s += v * asrc[hh * D_H + j]; d += v * adst[hh * D_H + j]; }
  ls[i] = s; ld[i] = d;
}

// pass 1: leaky-relu logit per (edge, head), scatter-max into m[dst]
__global__ void k_gat_edge_logit(const int* __restrict__ src, const int* __restrict__ dst,
                                 const float* __restrict__ ls, const float* __restrict__ ld,
                                 float* __restrict__ wbuf, float* __restrict__ mbuf,
                                 int nE, int nTot) {
  int t = blockIdx.x * blockDim.x + threadIdx.x;
  if (t >= nTot * N_HEAD) return;
  int e = t >> 2, hh = t & 3;
  int s, d;
  if (e < nE) { s = src[e]; d = dst[e]; } else { s = d = e - nE; }
  float l = ls[s * N_HEAD + hh] + ld[d * N_HEAD + hh];
  l = (l > 0.f) ? l : 0.2f * l;
  wbuf[t] = l;
  atomMaxF(&mbuf[d * N_HEAD + hh], l);
}

// pass 2: w = exp(logit - m[dst]); scatter-sum into z[dst]
__global__ void k_edge_exp(const int* __restrict__ dst, const float* __restrict__ mbuf,
                           float* __restrict__ wbuf, float* __restrict__ zbuf,
                           int nE, int nTot) {
  int t = blockIdx.x * blockDim.x + threadIdx.x;
  if (t >= nTot * N_HEAD) return;
  int e = t >> 2, hh = t & 3;
  int d = (e < nE) ? dst[e] : (e - nE);
  float w = __expf(wbuf[t] - mbuf[d * N_HEAD + hh]);
  wbuf[t] = w;
  atomAddF(&zbuf[d * N_HEAD + hh], w);
}

// pass 3: acc[dst,h,:] += alpha * feat[src,h,:]  (one wave per (edge,head))
__global__ void k_edge_scatter(const int* __restrict__ src, const int* __restrict__ dst,
                               const float* __restrict__ feat, const float* __restrict__ wbuf,
                               const float* __restrict__ zbuf, float* __restrict__ acc,
                               int nE, int nTot) {
  size_t t = (size_t)blockIdx.x * blockDim.x + threadIdx.x;
  int lane = (int)(t & 31);
  size_t eh = t >> 5;
  if (eh >= (size_t)nTot * N_HEAD) return;
  int e = (int)(eh >> 2), hh = (int)(eh & 3);
  int s, d;
  if (e < nE) { s = src[e]; d = dst[e]; } else { s = d = e - nE; }
  float alpha = wbuf[eh] / (zbuf[d * N_HEAD + hh] + 1e-16f);
  const float* f = feat + (size_t)s * HD + hh * D_H;
  float* a = acc + (size_t)d * HD + hh * D_H;
  atomAddF(&a[lane],      f[lane]      * alpha);
  atomAddF(&a[lane + 32], f[lane + 32] * alpha);
}

// mean over heads (+optional bias, relu), optional bf16 mirror
__global__ void k_head_mean(const float* __restrict__ acc, const float* __restrict__ bias,
                            float* __restrict__ out, __bf16* __restrict__ obf, int relu) {
  int i = blockIdx.x * blockDim.x + threadIdx.x;
  if (i >= N_NODES * D_H) return;
  int nd = i >> 6, dd = i & 63;
  const float* a = acc + (size_t)nd * HD;
  float v = 0.25f * (a[dd] + a[D_H + dd] + a[2 * D_H + dd] + a[3 * D_H + dd]);
  if (bias) v += bias[dd];
  if (relu) v = fmaxf(v, 0.f);
  out[i] = v;
  if (obf) obf[i] = (__bf16)v;
}

// ---------------- generic 64-wide scatter add (GIN / SAGE) -----------------
__global__ void k_scatter_add64(const int* __restrict__ src, const int* __restrict__ dst,
                                const float* __restrict__ x, float* __restrict__ out, int nE) {
  size_t t = (size_t)blockIdx.x * blockDim.x + threadIdx.x;
  int lane = (int)(t & 31);
  size_t e = t >> 5;
  if (e >= (size_t)nE) return;
  int s = src[e], d = dst[e];
  const float* f = x + (size_t)s * D_H;
  float* o = out + (size_t)d * D_H;
  atomAddF(&o[lane],      f[lane]);
  atomAddF(&o[lane + 32], f[lane + 32]);
}

__global__ void k_add_bf(const float* __restrict__ a, const float* __restrict__ b,
                         __bf16* __restrict__ o, size_t n) {
  size_t i = (size_t)blockIdx.x * blockDim.x + threadIdx.x;
  if (i < n) o[i] = (__bf16)(a[i] + b[i]);
}

__global__ void k_deg(const int* __restrict__ dst, float* __restrict__ deg, int nE) {
  int e = blockIdx.x * blockDim.x + threadIdx.x;
  if (e < nE) atomAddF(&deg[dst[e]], 1.f);
}

__global__ void k_div_deg(const float* __restrict__ s, const float* __restrict__ deg,
                          __bf16* __restrict__ o, size_t n) {
  size_t i = (size_t)blockIdx.x * blockDim.x + threadIdx.x;
  if (i < n) o[i] = (__bf16)(s[i] / fmaxf(deg[i >> 6], 1.f));
}

// ---------------- TransformerConv logit ------------------------------------
__global__ void k_tr_logit(const int* __restrict__ src, const int* __restrict__ dst,
                           const float* __restrict__ q, const float* __restrict__ k,
                           float* __restrict__ wbuf, float* __restrict__ mbuf, int nE) {
  int t = blockIdx.x * blockDim.x + threadIdx.x;
  if (t >= nE * N_HEAD) return;
  int e = t >> 2, hh = t & 3;
  int s = src[e], d = dst[e];
  const float4* qp = (const float4*)(q + (size_t)d * HD + hh * D_H);
  const float4* kp = (const float4*)(k + (size_t)s * HD + hh * D_H);
  float acc = 0.f;
#pragma unroll
  for (int j = 0; j < 16; ++j) {
    float4 a = qp[j], b = kp[j];
    acc += a.x * b.x + a.y * b.y + a.z * b.z + a.w * b.w;
  }
  float l = acc * 0.125f;   // 1/sqrt(64)
  wbuf[t] = l;
  atomMaxF(&mbuf[d * N_HEAD + hh], l);
}

// ---------------- EdgeConv: fused 2-layer MLP + segment max ----------------
// One wave per 16-edge tile. feat = [x_i, x_j - x_i] built on the fly,
// GEMM1 (K=128) -> relu -> LDS (bf16) -> GEMM2 (K=64) -> atomic max.
__device__ __forceinline__ v8bf ec_feat8(const float* xd, const float* xs, int off) {
  v8bf r;
  if (off < D_H) {
    const float4* p = (const float4*)(xd + off);
    float4 f0 = p[0], f1 = p[1];
    r[0]=(__bf16)f0.x; r[1]=(__bf16)f0.y; r[2]=(__bf16)f0.z; r[3]=(__bf16)f0.w;
    r[4]=(__bf16)f1.x; r[5]=(__bf16)f1.y; r[6]=(__bf16)f1.z; r[7]=(__bf16)f1.w;
  } else {
    int o = off - D_H;
    const float4* ps = (const float4*)(xs + o);
    const float4* pd = (const float4*)(xd + o);
    float4 a0 = ps[0], a1 = ps[1], b0 = pd[0], b1 = pd[1];
    r[0]=(__bf16)(a0.x-b0.x); r[1]=(__bf16)(a0.y-b0.y);
    r[2]=(__bf16)(a0.z-b0.z); r[3]=(__bf16)(a0.w-b0.w);
    r[4]=(__bf16)(a1.x-b1.x); r[5]=(__bf16)(a1.y-b1.y);
    r[6]=(__bf16)(a1.z-b1.z); r[7]=(__bf16)(a1.w-b1.w);
  }
  return r;
}

__global__ void k_edgeconv(const int* __restrict__ src, const int* __restrict__ dst,
                           const float* __restrict__ x5,
                           const __bf16* __restrict__ W1t, const float* __restrict__ b1,
                           const __bf16* __restrict__ W2t, const float* __restrict__ b2,
                           float* __restrict__ emax) {
  __shared__ __bf16 lds[8 * 16 * D_H];
  int widx = threadIdx.x >> 5;
  int lane = threadIdx.x & 31;
  int tile = blockIdx.x * 8 + widx;
  int e0 = tile << 4;
  int ll  = lane & 15;
  int sel = lane >> 4;
  int e = e0 + ll;
  int sN = src[e], dN = dst[e];
  const float* xd = x5 + (size_t)dN * D_H;
  const float* xs = x5 + (size_t)sN * D_H;

  // GEMM1: feat[16,128] @ ec_W1 -> [16,64]
  v8f acc1[4] = {};
#pragma unroll
  for (int k0 = 0; k0 < 2 * D_H; k0 += 32) {
    int olo = k0 + sel * 8;
    int ohi = k0 + 16 + sel * 8;
    v8bf alo = ec_feat8(xd, xs, olo);
    v8bf ahi = ec_feat8(xd, xs, ohi);
    v16bf af = __builtin_shufflevector(alo, ahi, 0,1,2,3,4,5,6,7,8,9,10,11,12,13,14,15);
#pragma unroll
    for (int c = 0; c < 4; ++c) {
      const __bf16* brow = W1t + (size_t)((c << 4) + ll) * (2 * D_H);
      v8bf blo = *(const v8bf*)(brow + olo);
      v8bf bhi = *(const v8bf*)(brow + ohi);
      v16bf bfr = __builtin_shufflevector(blo, bhi, 0,1,2,3,4,5,6,7,8,9,10,11,12,13,14,15);
      acc1[c] = __builtin_amdgcn_wmma_f32_16x16x32_bf16(false, af, false, bfr,
                                                        (short)0, acc1[c], false, false);
    }
  }
  // relu + bias, stage to LDS as bf16 [16][64]
  __bf16* my = lds + widx * (16 * D_H);
#pragma unroll
  for (int c = 0; c < 4; ++c) {
    int n = (c << 4) + ll;
    float bb = b1[n];
#pragma unroll
    for (int r = 0; r < 8; ++r) {
      int m = r + sel * 8;
      float v = fmaxf(acc1[c][r] + bb, 0.f);
      my[m * D_H + n] = (__bf16)v;
    }
  }
  __syncthreads();

  // GEMM2: [16,64] @ ec_W2 -> [16,64]
  v8f acc2[4] = {};
  const __bf16* arow = my + ll * D_H;
#pragma unroll
  for (int k0 = 0; k0 < D_H; k0 += 32) {
    v8bf alo = *(const v8bf*)(arow + k0 + sel * 8);
    v8bf ahi = *(const v8bf*)(arow + k0 + 16 + sel * 8);
    v16bf af = __builtin_shufflevector(alo, ahi, 0,1,2,3,4,5,6,7,8,9,10,11,12,13,14,15);
#pragma unroll
    for (int c = 0; c < 4; ++c) {
      const __bf16* brow = W2t + (size_t)((c << 4) + ll) * D_H;
      v8bf blo = *(const v8bf*)(brow + k0 + sel * 8);
      v8bf bhi = *(const v8bf*)(brow + k0 + 16 + sel * 8);
      v16bf bfr = __builtin_shufflevector(blo, bhi, 0,1,2,3,4,5,6,7,8,9,10,11,12,13,14,15);
      acc2[c] = __builtin_amdgcn_wmma_f32_16x16x32_bf16(false, af, false, bfr,
                                                        (short)0, acc2[c], false, false);
    }
  }
  // segment max into emax[dst]
#pragma unroll
  for (int c = 0; c < 4; ++c) {
    int n = (c << 4) + ll;
    float bb = b2[n];
#pragma unroll
    for (int r = 0; r < 8; ++r) {
      int er = e0 + r + sel * 8;
      int dn = dst[er];
      atomMaxF(&emax[(size_t)dn * D_H + n], acc2[c][r] + bb);
    }
  }
}

__global__ void k_x6max(const float* __restrict__ emax, float* __restrict__ x6, size_t n) {
  size_t i = (size_t)blockIdx.x * blockDim.x + threadIdx.x;
  if (i < n) x6[i] = fmaxf(emax[i], 0.f);   // -inf (isolated) -> 0, then relu
}

// ---------------- pooling ---------------------------------------------------
__global__ void k_cnt(const int* __restrict__ batch, float* __restrict__ cnt) {
  int i = blockIdx.x * blockDim.x + threadIdx.x;
  if (i < N_NODES) atomAddF(&cnt[batch[i]], 1.f);
}
__global__ void k_gsum(const int* __restrict__ batch, const float* __restrict__ x6,
                       float* __restrict__ gsum) {
  size_t t = (size_t)blockIdx.x * blockDim.x + threadIdx.x;
  int lane = (int)(t & 31);
  size_t n = t >> 5;
  if (n >= N_NODES) return;
  int b = batch[n];
  atomAddF(&gsum[b * D_H + lane],      x6[n * D_H + lane]);
  atomAddF(&gsum[b * D_H + lane + 32], x6[n * D_H + lane + 32]);
}
__global__ void k_gmean(const float* __restrict__ gsum, const float* __restrict__ cnt,
                        __bf16* __restrict__ o) {
  int i = blockIdx.x * blockDim.x + threadIdx.x;
  if (i >= N_G * D_H) return;
  o[i] = (__bf16)(gsum[i] / fmaxf(cnt[i >> 6], 1.f));
}

// ===========================================================================
extern "C" void kernel_launch(void* const* d_in, const int* in_sizes, int n_in,
                              void* d_out, int out_size, void* d_ws, size_t ws_size,
                              hipStream_t stream) {
  (void)in_sizes; (void)n_in; (void)out_size; (void)ws_size;
  const float* x         = (const float*)d_in[0];
  const int*   ei        = (const int*)d_in[1];
  const int*   batch     = (const int*)d_in[2];
  const float* gat1_W    = (const float*)d_in[3];
  const float* gat1_asrc = (const float*)d_in[4];
  const float* gat1_adst = (const float*)d_in[5];
  const float* gat1_b    = (const float*)d_in[6];
  const float* gat2_W    = (const float*)d_in[7];
  const float* gat2_asrc = (const float*)d_in[8];
  const float* gat2_adst = (const float*)d_in[9];
  const float* gat2_b    = (const float*)d_in[10];
  const float* gin_W1    = (const float*)d_in[11];
  const float* gin_b1    = (const float*)d_in[12];
  const float* gin_W2    = (const float*)d_in[13];
  const float* gin_b2    = (const float*)d_in[14];
  const float* sage1_Wl  = (const float*)d_in[15];
  const float* sage1_bl  = (const float*)d_in[16];
  const float* sage1_Wr  = (const float*)d_in[17];
  const float* sage2_Wl  = (const float*)d_in[18];
  const float* sage2_bl  = (const float*)d_in[19];
  const float* sage2_Wr  = (const float*)d_in[20];
  const float* tr_Wq     = (const float*)d_in[21];
  const float* tr_bq     = (const float*)d_in[22];
  const float* tr_Wk     = (const float*)d_in[23];
  const float* tr_bk     = (const float*)d_in[24];
  const float* tr_Wv     = (const float*)d_in[25];
  const float* tr_bv     = (const float*)d_in[26];
  const float* tr_Ws     = (const float*)d_in[27];
  const float* tr_bs     = (const float*)d_in[28];
  const float* ec_W1     = (const float*)d_in[29];
  const float* ec_b1     = (const float*)d_in[30];
  const float* ec_W2     = (const float*)d_in[31];
  const float* ec_b2     = (const float*)d_in[32];
  const float* fc1_W     = (const float*)d_in[33];
  const float* fc1_b     = (const float*)d_in[34];
  const float* fc2_W     = (const float*)d_in[35];
  const float* fc2_b     = (const float*)d_in[36];
  const int* src = ei;
  const int* dst = ei + N_EDGES;

  // ---- workspace carve ----
  char* base = (char*)d_ws;
  size_t off = 0;
  auto alloc = [&](size_t bytes) -> void* {
    void* p = base + off;
    off += (bytes + 255) & ~(size_t)255;
    return p;
  };
  __bf16* xbf       = (__bf16*)alloc((size_t)N_NODES * D_IN * 2);
  __bf16* gat1_Wt   = (__bf16*)alloc((size_t)HD * D_IN * 2);
  __bf16* gat2_Wt   = (__bf16*)alloc((size_t)HD * D_H * 2);
  __bf16* gin_W1t   = (__bf16*)alloc((size_t)D_H * D_H * 2);
  __bf16* gin_W2t   = (__bf16*)alloc((size_t)D_H * D_H * 2);
  __bf16* sage1_Wlt = (__bf16*)alloc((size_t)D_H * D_H * 2);
  __bf16* sage1_Wrt = (__bf16*)alloc((size_t)D_H * D_H * 2);
  __bf16* sage2_Wlt = (__bf16*)alloc((size_t)D_H * D_H * 2);
  __bf16* sage2_Wrt = (__bf16*)alloc((size_t)D_H * D_H * 2);
  __bf16* tr_Wqt    = (__bf16*)alloc((size_t)HD * D_H * 2);
  __bf16* tr_Wkt    = (__bf16*)alloc((size_t)HD * D_H * 2);
  __bf16* tr_Wvt    = (__bf16*)alloc((size_t)HD * D_H * 2);
  __bf16* tr_Wst    = (__bf16*)alloc((size_t)D_H * D_H * 2);
  __bf16* ec_W1t    = (__bf16*)alloc((size_t)D_H * (2 * D_H) * 2);
  __bf16* ec_W2t    = (__bf16*)alloc((size_t)D_H * D_H * 2);
  __bf16* fc1_Wt    = (__bf16*)alloc((size_t)D_H * D_H * 2);
  __bf16* fc2_Wt    = (__bf16*)alloc((size_t)D_H * D_H * 2);

  float* hbuf  = (float*)alloc((size_t)N_NODES * HD * 4);
  float* accb  = (float*)alloc((size_t)N_NODES * HD * 4);
  float* qb    = (float*)alloc((size_t)N_NODES * HD * 4);
  float* kb    = (float*)alloc((size_t)N_NODES * HD * 4);
  float* vb    = (float*)alloc((size_t)N_NODES * HD * 4);
  float* lsb   = (float*)alloc((size_t)N_NODES * N_HEAD * 4);
  float* ldb   = (float*)alloc((size_t)N_NODES * N_HEAD * 4);
  float* mb    = (float*)alloc((size_t)N_NODES * N_HEAD * 4);
  float* zb    = (float*)alloc((size_t)N_NODES * N_HEAD * 4);
  float* wb    = (float*)alloc((size_t)E_SL * N_HEAD * 4);
  float* degb  = (float*)alloc((size_t)N_NODES * 4);

  float*  xA   = (float*)alloc((size_t)N_NODES * D_H * 4);   // x1/x3/x4/x6 slots
  float*  xB   = (float*)alloc((size_t)N_NODES * D_H * 4);   // x2/xs/x5 slots
  float*  tmpf = (float*)alloc((size_t)N_NODES * D_H * 4);   // agg / sum / tmean
  float*  tmp2 = (float*)alloc((size_t)N_NODES * D_H * 4);
  __bf16* bfA  = (__bf16*)alloc((size_t)N_NODES * D_H * 2);
  __bf16* bfB  = (__bf16*)alloc((size_t)N_NODES * D_H * 2);
  __bf16* bfT  = (__bf16*)alloc((size_t)N_NODES * D_H * 2);

  float*  gsum  = (float*)alloc((size_t)N_G * D_H * 4);
  float*  cnt   = (float*)alloc((size_t)N_G * 4);
  __bf16* gmb   = (__bf16*)alloc((size_t)N_G * D_H * 2);
  float*  fct   = (float*)alloc((size_t)N_G * D_H * 4);
  __bf16* fctbf = (__bf16*)alloc((size_t)N_G * D_H * 2);

  const float NEG_INF = -std::numeric_limits<float>::infinity();
  auto NB = [](size_t n) { return (unsigned)((n + 255) / 256); };
  auto gemm = [&](const __bf16* A, const __bf16* Wt, const float* bias, const float* Cadd,
                  float* C, __bf16* Cbf, int M, int K, int Nout, int relu) {
    int tiles = (M / 16) * (Nout / 64);
    k_gemm<<<dim3((tiles + 7) / 8), dim3(256), 0, stream>>>(A, Wt, bias, Cadd, C, Cbf,
                                                            M, K, Nout, relu);
  };

  // ---- prep: bf16 x, transposed bf16 weights ----
  k_f2bf<<<NB((size_t)N_NODES * D_IN), 256, 0, stream>>>(x, xbf, (size_t)N_NODES * D_IN);
  k_wT<<<NB(D_IN * HD), 256, 0, stream>>>(gat1_W, gat1_Wt, D_IN, HD);
  k_wT<<<NB(D_H * HD), 256, 0, stream>>>(gat2_W, gat2_Wt, D_H, HD);
  k_wT<<<NB(D_H * D_H), 256, 0, stream>>>(gin_W1, gin_W1t, D_H, D_H);
  k_wT<<<NB(D_H * D_H), 256, 0, stream>>>(gin_W2, gin_W2t, D_H, D_H);
  k_wT<<<NB(D_H * D_H), 256, 0, stream>>>(sage1_Wl, sage1_Wlt, D_H, D_H);
  k_wT<<<NB(D_H * D_H), 256, 0, stream>>>(sage1_Wr, sage1_Wrt, D_H, D_H);
  k_wT<<<NB(D_H * D_H), 256, 0, stream>>>(sage2_Wl, sage2_Wlt, D_H, D_H);
  k_wT<<<NB(D_H * D_H), 256, 0, stream>>>(sage2_Wr, sage2_Wrt, D_H, D_H);
  k_wT<<<NB(D_H * HD), 256, 0, stream>>>(tr_Wq, tr_Wqt, D_H, HD);
  k_wT<<<NB(D_H * HD), 256, 0, stream>>>(tr_Wk, tr_Wkt, D_H, HD);
  k_wT<<<NB(D_H * HD), 256, 0, stream>>>(tr_Wv, tr_Wvt, D_H, HD);
  k_wT<<<NB(D_H * D_H), 256, 0, stream>>>(tr_Ws, tr_Wst, D_H, D_H);
  k_wT<<<NB(2 * D_H * D_H), 256, 0, stream>>>(ec_W1, ec_W1t, 2 * D_H, D_H);
  k_wT<<<NB(D_H * D_H), 256, 0, stream>>>(ec_W2, ec_W2t, D_H, D_H);
  k_wT<<<NB(D_H * D_H), 256, 0, stream>>>(fc1_W, fc1_Wt, D_H, D_H);
  k_wT<<<NB(D_H * D_H), 256, 0, stream>>>(fc2_W, fc2_Wt, D_H, D_H);

  // ---- GAT layer (shared sequence) ----
  auto gat_layer = [&](const __bf16* inbf, int Kin, const __bf16* Wt,
                       const float* asrc, const float* adst, const float* bias,
                       float* xout, __bf16* xoutbf) {
    gemm(inbf, Wt, nullptr, nullptr, hbuf, nullptr, N_NODES, Kin, HD, 0);
    k_attn_logits<<<NB(N_NODES * N_HEAD), 256, 0, stream>>>(hbuf, asrc, adst, lsb, ldb);
    k_fill<<<NB((size_t)N_NODES * N_HEAD), 256, 0, stream>>>(mb, NEG_INF, (size_t)N_NODES * N_HEAD);
    k_fill<<<NB((size_t)N_NODES * N_HEAD), 256, 0, stream>>>(zb, 0.f, (size_t)N_NODES * N_HEAD);
    k_fill<<<NB((size_t)N_NODES * HD), 256, 0, stream>>>(accb, 0.f, (size_t)N_NODES * HD);
    k_gat_edge_logit<<<NB((size_t)E_SL * N_HEAD), 256, 0, stream>>>(src, dst, lsb, ldb, wb, mb,
                                                                    N_EDGES, E_SL);
    k_edge_exp<<<NB((size_t)E_SL * N_HEAD), 256, 0, stream>>>(dst, mb, wb, zb, N_EDGES, E_SL);
    k_edge_scatter<<<NB((size_t)E_SL * N_HEAD * 32), 256, 0, stream>>>(src, dst, hbuf, wb, zb,
                                                                       accb, N_EDGES, E_SL);
    k_head_mean<<<NB((size_t)N_NODES * D_H), 256, 0, stream>>>(accb, bias, xout, xoutbf, 1);
  };

  gat_layer(xbf, D_IN, gat1_Wt, gat1_asrc, gat1_adst, gat1_b, xA, bfA);   // x1
  gat_layer(bfA, D_H, gat2_Wt, gat2_asrc, gat2_adst, gat2_b, xB, bfB);    // x2

  // ---- GIN ----
  k_fill<<<NB((size_t)N_NODES * D_H), 256, 0, stream>>>(tmpf, 0.f, (size_t)N_NODES * D_H);
  k_scatter_add64<<<NB((size_t)N_EDGES * 32), 256, 0, stream>>>(src, dst, xB, tmpf, N_EDGES);
  k_add_bf<<<NB((size_t)N_NODES * D_H), 256, 0, stream>>>(xB, tmpf, bfT, (size_t)N_NODES * D_H);
  gemm(bfT, gin_W1t, gin_b1, nullptr, tmp2, bfB, N_NODES, D_H, D_H, 1);   // t1 (reuse bfB)
  gemm(bfB, gin_W2t, gin_b2, nullptr, xA, bfA, N_NODES, D_H, D_H, 1);     // x3

  // ---- degrees (shared by both SAGE layers) ----
  k_fill<<<NB((size_t)N_NODES), 256, 0, stream>>>(degb, 0.f, (size_t)N_NODES);
  k_deg<<<NB((size_t)N_EDGES), 256, 0, stream>>>(dst, degb, N_EDGES);

  // ---- SAGE1: xs = relu(mean@Wl + bl + x3@Wr) ----
  k_fill<<<NB((size_t)N_NODES * D_H), 256, 0, stream>>>(tmpf, 0.f, (size_t)N_NODES * D_H);
  k_scatter_add64<<<NB((size_t)N_EDGES * 32), 256, 0, stream>>>(src, dst, xA, tmpf, N_EDGES);
  k_div_deg<<<NB((size_t)N_NODES * D_H), 256, 0, stream>>>(tmpf, degb, bfT, (size_t)N_NODES * D_H);
  gemm(bfT, sage1_Wlt, sage1_bl, nullptr, tmp2, nullptr, N_NODES, D_H, D_H, 0);
  gemm(bfA, sage1_Wrt, nullptr, tmp2, xB, bfB, N_NODES, D_H, D_H, 1);     // xs

  // ---- SAGE2: x4 ----
  k_fill<<<NB((size_t)N_NODES * D_H), 256, 0, stream>>>(tmpf, 0.f, (size_t)N_NODES * D_H);
  k_scatter_add64<<<NB((size_t)N_EDGES * 32), 256, 0, stream>>>(src, dst, xB, tmpf, N_EDGES);
  k_div_deg<<<NB((size_t)N_NODES * D_H), 256, 0, stream>>>(tmpf, degb, bfT, (size_t)N_NODES * D_H);
  gemm(bfT, sage2_Wlt, sage2_bl, nullptr, tmp2, nullptr, N_NODES, D_H, D_H, 0);
  gemm(bfB, sage2_Wrt, nullptr, tmp2, xA, bfA, N_NODES, D_H, D_H, 1);     // x4 (xA/bfA)

  // ---- TransformerConv ----
  gemm(bfA, tr_Wqt, tr_bq, nullptr, qb, nullptr, N_NODES, D_H, HD, 0);
  gemm(bfA, tr_Wkt, tr_bk, nullptr, kb, nullptr, N_NODES, D_H, HD, 0);
  gemm(bfA, tr_Wvt, tr_bv, nullptr, vb, nullptr, N_NODES, D_H, HD, 0);
  k_fill<<<NB((size_t)N_NODES * N_HEAD), 256, 0, stream>>>(mb, NEG_INF, (size_t)N_NODES * N_HEAD);
  k_fill<<<NB((size_t)N_NODES * N_HEAD), 256, 0, stream>>>(zb, 0.f, (size_t)N_NODES * N_HEAD);
  k_fill<<<NB((size_t)N_NODES * HD), 256, 0, stream>>>(accb, 0.f, (size_t)N_NODES * HD);
  k_tr_logit<<<NB((size_t)N_EDGES * N_HEAD), 256, 0, stream>>>(src, dst, qb, kb, wb, mb, N_EDGES);
  k_edge_exp<<<NB((size_t)N_EDGES * N_HEAD), 256, 0, stream>>>(dst, mb, wb, zb, N_EDGES, N_EDGES);
  k_edge_scatter<<<NB((size_t)N_EDGES * N_HEAD * 32), 256, 0, stream>>>(src, dst, vb, wb, zb,
                                                                        accb, N_EDGES, N_EDGES);
  k_head_mean<<<NB((size_t)N_NODES * D_H), 256, 0, stream>>>(accb, nullptr, tmpf, nullptr, 0);
  gemm(bfA, tr_Wst, tr_bs, tmpf, xB, nullptr, N_NODES, D_H, D_H, 1);      // x5 (f32 only)

  // ---- EdgeConv (fused WMMA MLP + segment max) ----
  k_fill<<<NB((size_t)N_NODES * D_H), 256, 0, stream>>>(tmp2, NEG_INF, (size_t)N_NODES * D_H);
  k_edgeconv<<<dim3(N_EDGES / 16 / 8), dim3(256), 0, stream>>>(src, dst, xB, ec_W1t, ec_b1,
                                                               ec_W2t, ec_b2, tmp2);
  k_x6max<<<NB((size_t)N_NODES * D_H), 256, 0, stream>>>(tmp2, xA, (size_t)N_NODES * D_H);  // x6

  // ---- global mean pool + FC head ----
  k_fill<<<1, 256, 0, stream>>>(gsum, 0.f, (size_t)N_G * D_H);
  k_fill<<<1, 256, 0, stream>>>(cnt, 0.f, (size_t)N_G);
  k_cnt<<<NB((size_t)N_NODES), 256, 0, stream>>>(batch, cnt);
  k_gsum<<<NB((size_t)N_NODES * 32), 256, 0, stream>>>(batch, xA, gsum);
  k_gmean<<<NB((size_t)N_G * D_H), 256, 0, stream>>>(gsum, cnt, gmb);
  gemm(gmb, fc1_Wt, fc1_b, nullptr, fct, fctbf, N_G, D_H, D_H, 1);
  gemm(fctbf, fc2_Wt, fc2_b, nullptr, (float*)d_out, nullptr, N_G, D_H, D_H, 0);
}
